// PICInputNet_89979564851452
// MI455X (gfx1250) — matrix operand order repaired and checked by reading, not compiled, not run on hardware
//
#include <hip/hip_runtime.h>
#include <hip/hip_fp16.h>

// ---------------------------------------------------------------------------
// PICInputNet fused kernel for MI455X (gfx1250, wave32, WMMA).
//
// Shapes: G=784 groups, per group: tanh(W1_g[64x64] @ feat^T[64x2048]) then
// W2_g[64x64] @ h[64x2048], output (g, q, p) fp32 = 411 MB (the roofline
// bottleneck: ~17.6us @ 23.3 TB/s). 26.3 GFLOP of GEMM -> needs matrix core
// (v_wmma_f32_16x16x32_f16) to stay store-bandwidth-bound. Intermediate h
// (another 411MB) is fused away through registers + LDS.
// ---------------------------------------------------------------------------

typedef __attribute__((ext_vector_type(16))) _Float16 v16h;
typedef __attribute__((ext_vector_type(8)))  _Float16 v8h;
typedef __attribute__((ext_vector_type(8)))  float    v8f;
typedef __attribute__((ext_vector_type(4)))  float    v4f;

#define QTOT   2048
#define GROUPS 784
#define NETD   64
#define NPAR   64
#define FFD    64
#define QT     16          // q columns per wave tile
#define NTILES (QTOT/QT)   // 128

// ---------------------------------------------------------------------------
// Kernel 1: Fourier features, written directly in WMMA B-operand layout.
// B operand (32x16, KxN) in this code's contiguous k order (A uses the same), wave32:
//   lane L: N = L&15 ; lanes 0-15 hold K=kstep*32+0..15, lanes 16-31 hold
//   K=kstep*32+16..31, packed as 16 contiguous halfs per lane.
// Flat ws index for (qtile t, kstep s, lane L) = ((t*2+s)*32 + L) * 16 halfs.
// Total: 128*2*32*32B = 256 KB in d_ws (resident in L2 for the main kernel).
// ---------------------------------------------------------------------------
__global__ void feat_kernel(const float* __restrict__ z,
                            const float* __restrict__ coeff,
                            _Float16* __restrict__ featws) {
  const int idx  = blockIdx.x * blockDim.x + threadIdx.x;  // 0..8191
  const int lane = idx & 31;
  const int s    = (idx >> 5) & 1;
  const int t    = idx >> 6;
  const int q    = t * QT + (lane & 15);
  const int f0   = s * 32 + ((lane >> 4) << 4);
  const float zq = 6.283185307179586f * z[q];
  v16h out;
#pragma unroll
  for (int j = 0; j < 16; ++j) {
    const int f   = f0 + j;
    const float c = coeff[(f < 32) ? f : (f - 32)];
    const float x = zq * c;
    out[j] = (_Float16)((f < 32) ? __cosf(x) : __sinf(x));
  }
  *(v16h*)(featws + (size_t)idx * 16) = out;
}

// ---------------------------------------------------------------------------
// Kernel 2: one workgroup (256 threads = 8 waves) per group g.
// ---------------------------------------------------------------------------
__global__ void __launch_bounds__(256)
picnet_kernel(const float* __restrict__ W1, const float* __restrict__ W2,
              const _Float16* __restrict__ featws, float* __restrict__ out) {
  __shared__ __align__(16) _Float16 sW1[NETD * FFD];        // 8 KB
  __shared__ __align__(16) _Float16 sW2[NPAR * NETD];       // 8 KB
  __shared__ __align__(16) _Float16 sH[8][QT * NETD];       // 16 KB (per-wave h panel)
  __shared__ __align__(16) float    sO[8][QT * NPAR];       // 32 KB (per-wave out stage)

  const int g    = blockIdx.x;
  const int tid  = threadIdx.x;
  const int lane = tid & 31;
  const int wave = tid >> 5;

  // ---- Phase 0: load W1_g, W2_g, convert fp32 -> f16 into LDS -------------
  {
    const float* w1g = W1 + (size_t)g * NETD * FFD;   // row m = g*64+m of (50176,64)
    const float* w2g = W2 + (size_t)g * NPAR * NETD;  // W2[g][p][n]
    const int base = tid * 16;                        // 4096 elems / 256 threads
#pragma unroll
    for (int k = 0; k < 16; k += 4) {
      const v4f a = *(const v4f*)(w1g + base + k);
      const v4f b = *(const v4f*)(w2g + base + k);
#pragma unroll
      for (int j = 0; j < 4; ++j) {
        sW1[base + k + j] = (_Float16)a[j];
        sW2[base + k + j] = (_Float16)b[j];
      }
    }
  }
  __syncthreads();

  // ---- Preload WMMA A operands (reused for all 16 q-iterations) -----------
  // 16-bit A layout (matches B): lane L -> M = L&15, kbase = (L>>4)*16,
  // element j: K = s*32 + kbase + j  => two contiguous 16B runs.
  const int mrow  = lane & 15;
  const int kbase = (lane >> 4) << 4;
  v16h A1[4][2], A2[4][2];
#pragma unroll
  for (int mt = 0; mt < 4; ++mt) {
#pragma unroll
    for (int s = 0; s < 2; ++s) {
      const int m = mt * 16 + mrow;
      const int f = s * 32 + kbase;
      union { v16h v; v8h h[2]; } u1, u2;
      u1.h[0] = *(const v8h*)(sW1 + m * FFD + f);
      u1.h[1] = *(const v8h*)(sW1 + m * FFD + f + 8);
      u2.h[0] = *(const v8h*)(sW2 + m * NETD + f);
      u2.h[1] = *(const v8h*)(sW2 + m * NETD + f + 8);
      A1[mt][s] = u1.v;
      A2[mt][s] = u2.v;
    }
  }

  const int qloc = lane & 15;   // C/D layout: N = lane&15
  const int hi   = lane >> 4;   // C/D layout: M half select (+8)
  float* outg = out + (size_t)g * QTOT * NPAR;

  // ---- Main loop: 8 waves x 16 iterations cover all 128 q-tiles -----------
  for (int it = 0; it < NTILES / 8; ++it) {
    const int t = it * 8 + wave;

    // B1: Fourier features, pre-laid-out -> one 32B load per lane per kstep
    v16h B1[2];
    B1[0] = *(const v16h*)(featws + ((size_t)(t * 2 + 0) * 32 + lane) * 16);
    B1[1] = *(const v16h*)(featws + ((size_t)(t * 2 + 1) * 32 + lane) * 16);

    // GEMM1 (h = W1_g @ feat^T) + fast tanh, staged to per-wave LDS panel.
    // D layout: vgpr r, lane L -> (M = mt*16 + 8*hi + r, N = qloc).
#pragma unroll
    for (int mt = 0; mt < 4; ++mt) {
      v8f acc = {};
      acc = __builtin_amdgcn_wmma_f32_16x16x32_f16(false, A1[mt][0], false, B1[0],
                                                   (short)0, acc, false, false);
      acc = __builtin_amdgcn_wmma_f32_16x16x32_f16(false, A1[mt][1], false, B1[1],
                                                   (short)0, acc, false, false);
      v8h hv;
#pragma unroll
      for (int r = 0; r < 8; ++r) {
        // tanh(x) = 1 - 2/(exp(2x)+1)  (v_exp_f32 + v_rcp_f32)
        const float e = __expf(2.0f * acc[r]);
        hv[r] = (_Float16)(1.0f - 2.0f * __builtin_amdgcn_rcpf(e + 1.0f));
      }
      // sH[wave] layout: [q][n] with n contiguous -> 8 consecutive n per store
      *(v8h*)(&sH[wave][qloc * NETD + mt * 16 + hi * 8]) = hv;
    }
    __syncthreads();

    // B2: re-stripe h panel into B-operand layout (lane: N=qloc, K range by half)
    v16h B2[2];
    B2[0] = *(const v16h*)(&sH[wave][qloc * NETD + 0 * 32 + hi * 16]);
    B2[1] = *(const v16h*)(&sH[wave][qloc * NETD + 1 * 32 + hi * 16]);

    // GEMM2 (out = W2_g @ h), stage fp32 16q x 64p block into LDS
#pragma unroll
    for (int pt = 0; pt < 4; ++pt) {
      v8f acc = {};
      acc = __builtin_amdgcn_wmma_f32_16x16x32_f16(false, A2[pt][0], false, B2[0],
                                                   (short)0, acc, false, false);
      acc = __builtin_amdgcn_wmma_f32_16x16x32_f16(false, A2[pt][1], false, B2[1],
                                                   (short)0, acc, false, false);
      const v4f lo = {acc[0], acc[1], acc[2], acc[3]};
      const v4f hp = {acc[4], acc[5], acc[6], acc[7]};
      *(v4f*)(&sO[wave][qloc * NPAR + pt * 16 + hi * 8])     = lo;
      *(v4f*)(&sO[wave][qloc * NPAR + pt * 16 + hi * 8 + 4]) = hp;
    }
    __syncthreads();

    // Coalesced writeout: the 4KB [16q][64p] block is contiguous in out[g][q][p].
    // 8 x (32 lanes x 16B = 512B contiguous) global_store_b128 per wave.
    float* dst = outg + (size_t)t * QT * NPAR;
    const float* src = sO[wave];
#pragma unroll
    for (int c = 0; c < 8; ++c) {
      const int off = (c * 32 + lane) * 4;
      *(v4f*)(dst + off) = *(const v4f*)(src + off);
    }
    __syncthreads();
  }
}

// ---------------------------------------------------------------------------
// Launch. Inputs: z_quad(2048) f32, coeff(32) f32, W1(50176*64) f32,
// W2(784*64*64) f32. Output: (784, 2048, 64) f32. d_ws: 256 KB feature stash.
// ---------------------------------------------------------------------------
extern "C" void kernel_launch(void* const* d_in, const int* in_sizes, int n_in,
                              void* d_out, int out_size, void* d_ws, size_t ws_size,
                              hipStream_t stream) {
  const float* z     = (const float*)d_in[0];
  const float* coeff = (const float*)d_in[1];
  const float* W1    = (const float*)d_in[2];
  const float* W2    = (const float*)d_in[3];
  float* out         = (float*)d_out;
  _Float16* featws   = (_Float16*)d_ws;

  // 128 tiles * 2 ksteps * 32 lanes = 8192 threads
  feat_kernel<<<32, 256, 0, stream>>>(z, coeff, featws);
  picnet_kernel<<<GROUPS, 256, 0, stream>>>(W1, W2, featws, out);
}
